// RWKV7_18219251270217
// MI455X (gfx1250) — matrix operand order repaired and checked by reading, not compiled, not run on hardware
//
#include <hip/hip_runtime.h>
#include <hip/hip_bf16.h>
#include <math.h>

typedef __attribute__((ext_vector_type(16))) _Float16 v16h;
typedef __attribute__((ext_vector_type(8)))  _Float16 v8h;
typedef __attribute__((ext_vector_type(8)))  float    v8f;

#define RW_B 2
#define RW_T 2048
#define RW_C 2048
#define RW_H 32
#define RW_N 64
#define RW_M (RW_B*RW_T)            // 4096 tokens
#define GN_EPS 6.4e-4f              // 1e-5 * 8^2

#define SHUF16(lo, hi) __builtin_shufflevector((lo), (hi), 0,1,2,3,4,5,6,7,8,9,10,11,12,13,14,15)

__device__ __forceinline__ float sigf(float x) { return 1.f / (1.f + expf(-x)); }

// ---------------------------------------------------------------------------
// f16 WMMA GEMM (NT only): Out[M,N] f32 = A[M,K] f16 row-major x B[N,K] f16
// row-major (i.e. x @ W^T with W pre-converted/transposed to [N,K] f16).
// Block = 256 thr = 8 waves, wave grid 4(M) x 2(N); wave tile 32x32, 4 accums.
// Per K-step(32): 8 x global_load_b128 feed 4 v_wmma_f32_16x16x32_f16.
// A-fragment (ISA 16-bit A 16x32): lane halfs = A[row, kb+8*hi .. +8) then
// A[row, kb+16+8*hi .. +8).  B-fragment (32x16): same k order as A.
// C layout: VGPR g -> row g + 8*hi, col = lane&15.
// ---------------------------------------------------------------------------
__global__ __launch_bounds__(256) void gemm_wmma16(const _Float16* __restrict__ A,
                                                   const _Float16* __restrict__ Bm,
                                                   float* __restrict__ Out,
                                                   int M, int N, int K) {
  const int lane = threadIdx.x & 31;
  const int wv   = threadIdx.x >> 5;
  const int m0 = blockIdx.y * 128 + (wv >> 1) * 32;
  const int n0 = blockIdx.x * 64  + (wv & 1)  * 32;
  if (m0 + 32 > M || n0 + 32 > N) return;   // uniform per wave; EXEC stays all-ones
  const int hi  = lane >> 4;
  const int r15 = lane & 15;

  const _Float16* Ar0 = A + (size_t)(m0 + r15) * K;
  const _Float16* Ar1 = Ar0 + (size_t)16 * K;
  const _Float16* Bc0 = Bm + (size_t)(n0 + r15) * K + 8 * hi;
  const _Float16* Bc1 = Bm + (size_t)(n0 + 16 + r15) * K + 8 * hi;

  v8f acc00 = {}, acc01 = {}, acc10 = {}, acc11 = {};

  for (int kb = 0; kb < K; kb += 32) {
    __builtin_prefetch(Ar0 + kb + 128, 0, 1);
    v16h a0, a1, b0, b1;
    {
      v8h lo0 = *(const v8h*)(Ar0 + kb + 8 * hi);
      v8h hi0 = *(const v8h*)(Ar0 + kb + 16 + 8 * hi);
      a0 = SHUF16(lo0, hi0);
      v8h lo1 = *(const v8h*)(Ar1 + kb + 8 * hi);
      v8h hi1 = *(const v8h*)(Ar1 + kb + 16 + 8 * hi);
      a1 = SHUF16(lo1, hi1);
    }
    b0 = SHUF16(*(const v8h*)(Bc0 + kb), *(const v8h*)(Bc0 + kb + 16));
    b1 = SHUF16(*(const v8h*)(Bc1 + kb), *(const v8h*)(Bc1 + kb + 16));

    acc00 = __builtin_amdgcn_wmma_f32_16x16x32_f16(false, a0, false, b0, (short)0, acc00, false, false);
    acc01 = __builtin_amdgcn_wmma_f32_16x16x32_f16(false, a0, false, b1, (short)0, acc01, false, false);
    acc10 = __builtin_amdgcn_wmma_f32_16x16x32_f16(false, a1, false, b0, (short)0, acc10, false, false);
    acc11 = __builtin_amdgcn_wmma_f32_16x16x32_f16(false, a1, false, b1, (short)0, acc11, false, false);
  }
#pragma unroll
  for (int g = 0; g < 8; ++g) {
    const size_t ra = (size_t)(m0 + g + 8 * hi) * N;
    const size_t rb = (size_t)(m0 + 16 + g + 8 * hi) * N;
    Out[ra + n0 + r15]      = acc00[g];
    Out[ra + n0 + 16 + r15] = acc01[g];
    Out[rb + n0 + r15]      = acc10[g];
    Out[rb + n0 + 16 + r15] = acc11[g];
  }
}

// ---------------------------------------------------------------------------
// conversions
// ---------------------------------------------------------------------------
__global__ void k_cvt(const float* __restrict__ in, _Float16* __restrict__ out, int n) {
  int i = blockIdx.x * blockDim.x + threadIdx.x;
  if (i < n) out[i] = (_Float16)in[i];
}

// LDS-tiled transpose+convert: in[K,N] f32 row-major -> out[N,K] f16 row-major.
// Block (32,8), 32x32 tiles; K and N are multiples of 32 for all MLP weights.
__global__ __launch_bounds__(256) void k_tcvt(const float* __restrict__ in,
                                              _Float16* __restrict__ out,
                                              int K, int N) {
  __shared__ float tile[32][33];
  const int tx = threadIdx.x, ty = threadIdx.y;
  const int kb = blockIdx.y * 32, nb = blockIdx.x * 32;
#pragma unroll
  for (int j = 0; j < 4; ++j)
    tile[ty + j * 8][tx] = in[(size_t)(kb + ty + j * 8) * N + nb + tx];
  __syncthreads();
#pragma unroll
  for (int j = 0; j < 4; ++j)
    out[(size_t)(nb + ty + j * 8) * K + kb + tx] = (_Float16)tile[tx][ty + j * 8];
}

// ---------------------------------------------------------------------------
// elementwise kernels
// ---------------------------------------------------------------------------
// time_shift mixing: xx = shift(x)-x ; x? = (f16)(x + xx*m?)
__global__ void k_mix(const float* __restrict__ x,
                      const float* __restrict__ mr, const float* __restrict__ mw,
                      const float* __restrict__ mk, const float* __restrict__ mv,
                      const float* __restrict__ ma, const float* __restrict__ mg,
                      _Float16* __restrict__ xr, _Float16* __restrict__ xw,
                      _Float16* __restrict__ xk, _Float16* __restrict__ xv,
                      _Float16* __restrict__ xa, _Float16* __restrict__ xg,
                      int n) {
  int i = blockIdx.x * blockDim.x + threadIdx.x;
  if (i >= n) return;
  int c   = i % RW_C;
  int pos = i % (RW_T * RW_C);
  float xi = x[i];
  float prev = (pos >= RW_C) ? x[i - RW_C] : 0.f;
  float xx = prev - xi;
  xr[i] = (_Float16)(xi + xx * mr[c]); xw[i] = (_Float16)(xi + xx * mw[c]);
  xk[i] = (_Float16)(xi + xx * mk[c]); xv[i] = (_Float16)(xi + xx * mv[c]);
  xa[i] = (_Float16)(xi + xx * ma[c]); xg[i] = (_Float16)(xi + xx * mg[c]);
}

__global__ void k_tanh_h(const float* __restrict__ z, _Float16* __restrict__ out, int n) {
  int i = blockIdx.x * blockDim.x + threadIdx.x;
  if (i < n) out[i] = (_Float16)tanhf(z[i]);
}
__global__ void k_sig_h(const float* __restrict__ z, _Float16* __restrict__ out, int n) {
  int i = blockIdx.x * blockDim.x + threadIdx.x;
  if (i < n) out[i] = (_Float16)sigf(z[i]);
}
// z <- sigmoid(bias[c] + z)
__global__ void k_sigb(float* __restrict__ z, const float* __restrict__ bias, int n) {
  int i = blockIdx.x * blockDim.x + threadIdx.x;
  if (i < n) z[i] = sigf(bias[i % RW_C] + z[i]);
}
// z <- exp(-exp(-softplus(-(w0+z)) - 0.5))
__global__ void k_decay(float* __restrict__ z, const float* __restrict__ w0, int n) {
  int i = blockIdx.x * blockDim.x + threadIdx.x;
  if (i >= n) return;
  float u  = w0[i % RW_C] + z[i];
  float sp = fmaxf(-u, 0.f) + log1pf(expf(-fabsf(u)));  // softplus(-u), stable
  z[i] = expf(-expf(-sp - 0.5f));
}
// v <- v + (v_first - v) * sigmoid(v0[c] + zv)
__global__ void k_vmix(float* __restrict__ v, const float* __restrict__ vf,
                       const float* __restrict__ zv, const float* __restrict__ v0, int n) {
  int i = blockIdx.x * blockDim.x + threadIdx.x;
  if (i >= n) return;
  float s = sigf(v0[i % RW_C] + zv[i]);
  float vi = v[i];
  v[i] = vi + (vf[i] - vi) * s;
}

// ---------------------------------------------------------------------------
// Per-token prep: per-head L2-normalize kk = k*k_k, build k2, relayout scan
// inputs [B,T,C] -> [B,H,T,N]. Block per token, 256 thr, 8 chan/thread.
// ---------------------------------------------------------------------------
__global__ __launch_bounds__(256) void k_prep(
    const float* __restrict__ r, const float* __restrict__ k,
    const float* __restrict__ v, const float* __restrict__ dec,
    const float* __restrict__ asig,
    const float* __restrict__ k_k, const float* __restrict__ k_a,
    float* __restrict__ rr, float* __restrict__ kk2, float* __restrict__ vv,
    float* __restrict__ dd, float* __restrict__ aa, float* __restrict__ bb) {
  const int tok = blockIdx.x;
  const int b = tok / RW_T, t = tok % RW_T;
  const int tid = threadIdx.x;
  const int h = tid >> 3;
  __shared__ float part[256];
  float kkbuf[8];
  float ss = 0.f;
  const size_t src0 = (size_t)tok * RW_C + tid * 8;
#pragma unroll
  for (int i = 0; i < 8; ++i) {
    int c = tid * 8 + i;
    float kc = k[src0 + i] * k_k[c];
    kkbuf[i] = kc;
    ss += kc * kc;
  }
  part[tid] = ss;
  __syncthreads();
  float sum = 0.f;
#pragma unroll
  for (int j = 0; j < 8; ++j) sum += part[h * 8 + j];
  const float inv = 1.f / fmaxf(sqrtf(sum), 1e-12f);
  const size_t dst0 = ((size_t)(b * RW_H + h) * RW_T + t) * RW_N + (tid & 7) * 8;
#pragma unroll
  for (int i = 0; i < 8; ++i) {
    int c = tid * 8 + i;
    size_t s = src0 + i, d = dst0 + i;
    float kn = kkbuf[i] * inv;
    float a_ = asig[s];
    rr[d]  = r[s];
    vv[d]  = v[s];
    dd[d]  = dec[s];
    kk2[d] = k[s] * (1.f + (a_ - 1.f) * k_a[c]);
    aa[d]  = -kn;
    bb[d]  = kn * a_;
  }
}

// ---------------------------------------------------------------------------
// Sequential state scan: one block per (b,h); thread i owns row i of S.
// sab uses old S; y uses updated S (matches reference step()).
// ---------------------------------------------------------------------------
__global__ __launch_bounds__(64) void k_scan(
    const float* __restrict__ rr, const float* __restrict__ kk2,
    const float* __restrict__ vv, const float* __restrict__ dd,
    const float* __restrict__ aa, const float* __restrict__ bb,
    float* __restrict__ y) {
  const int bh = blockIdx.x;
  const int i  = threadIdx.x;
  __shared__ float lr[64], lk[64], lv[64], ld[64], la[64], lb[64];
  float S[64];
#pragma unroll
  for (int j = 0; j < 64; ++j) S[j] = 0.f;
  const size_t base = (size_t)bh * RW_T * RW_N;
  for (int t = 0; t < RW_T; ++t) {
    const size_t o = base + (size_t)t * RW_N + i;
    lr[i] = rr[o]; lk[i] = kk2[o]; lv[i] = vv[o];
    ld[i] = dd[o]; la[i] = aa[o];  lb[i] = bb[o];
    __syncthreads();
    float sab = 0.f;
#pragma unroll 16
    for (int j = 0; j < 64; ++j) sab += S[j] * la[j];
    const float vi = lv[i];
    float yv = 0.f;
#pragma unroll 16
    for (int j = 0; j < 64; ++j) {
      float s = S[j] * ld[j] + sab * lb[j] + vi * lk[j];
      S[j] = s;
      yv += s * lr[j];
    }
    y[o] = yv;
    __syncthreads();
  }
}

// ---------------------------------------------------------------------------
// GroupNorm + (r*k2*r_k).sum * v bonus + gate; emits token-major f16 y2.
// ---------------------------------------------------------------------------
__global__ __launch_bounds__(256) void k_post(
    const float* __restrict__ y, const float* __restrict__ rr,
    const float* __restrict__ kk2, const float* __restrict__ vv,
    const float* __restrict__ g, const float* __restrict__ r_k,
    const float* __restrict__ ln_g, const float* __restrict__ ln_b,
    _Float16* __restrict__ y2) {
  const int tok = blockIdx.x;
  const int b = tok / RW_T, t = tok % RW_T;
  const int tid = threadIdx.x;
  const int h = tid >> 3;
  __shared__ float ps[256], ps2[256], prk[256];
  const size_t d0 = ((size_t)(b * RW_H + h) * RW_T + t) * RW_N + (tid & 7) * 8;
  float s = 0.f, s2 = 0.f, srk = 0.f;
  float yb[8];
#pragma unroll
  for (int i = 0; i < 8; ++i) {
    int c = tid * 8 + i;
    float yv = y[d0 + i];
    yb[i] = yv;
    s += yv; s2 += yv * yv;
    srk += rr[d0 + i] * kk2[d0 + i] * r_k[c];
  }
  ps[tid] = s; ps2[tid] = s2; prk[tid] = srk;
  __syncthreads();
  float sum = 0.f, sumsq = 0.f, rk = 0.f;
#pragma unroll
  for (int j = 0; j < 8; ++j) {
    sum += ps[h * 8 + j]; sumsq += ps2[h * 8 + j]; rk += prk[h * 8 + j];
  }
  const float mu   = sum * (1.f / RW_N);
  const float var  = sumsq * (1.f / RW_N) - mu * mu;
  const float rstd = rsqrtf(var + GN_EPS);
  const size_t o0 = (size_t)tok * RW_C + tid * 8;
#pragma unroll
  for (int i = 0; i < 8; ++i) {
    int c = tid * 8 + i;
    float yn = (yb[i] - mu) * rstd * ln_g[c] + ln_b[c];
    yn += rk * vv[d0 + i];
    y2[o0 + i] = (_Float16)(yn * g[o0 + i]);
  }
}

// ---------------------------------------------------------------------------
extern "C" void kernel_launch(void* const* d_in, const int* in_sizes, int n_in,
                              void* d_out, int out_size, void* d_ws, size_t ws_size,
                              hipStream_t stream) {
  const size_t BTC = (size_t)RW_B * RW_T * RW_C;     // 8,388,608
  const size_t CC  = (size_t)RW_C * RW_C;            // 4,194,304
  const float* x   = (const float*)d_in[0];
  const float* vf  = (const float*)d_in[1];
  const float* mr  = (const float*)d_in[2];
  const float* mw  = (const float*)d_in[3];
  const float* mk  = (const float*)d_in[4];
  const float* mv  = (const float*)d_in[5];
  const float* ma  = (const float*)d_in[6];
  const float* mg  = (const float*)d_in[7];
  const float* w0  = (const float*)d_in[8];
  const float* w1  = (const float*)d_in[9];
  const float* w2  = (const float*)d_in[10];
  const float* a0  = (const float*)d_in[11];
  const float* a1  = (const float*)d_in[12];
  const float* a2  = (const float*)d_in[13];
  const float* v0  = (const float*)d_in[14];
  const float* v1  = (const float*)d_in[15];
  const float* v2  = (const float*)d_in[16];
  const float* g1  = (const float*)d_in[17];
  const float* g2  = (const float*)d_in[18];
  const float* k_k = (const float*)d_in[19];
  const float* k_a = (const float*)d_in[20];
  const float* r_k = (const float*)d_in[21];
  const float* W_r = (const float*)d_in[22];
  const float* W_k = (const float*)d_in[23];
  const float* W_v = (const float*)d_in[24];
  const float* W_o = (const float*)d_in[25];
  const float* lng = (const float*)d_in[26];
  const float* lnb = (const float*)d_in[27];
  float* out = (float*)d_out;

  // ---- workspace carve (float units) ----
  float* f = (float*)d_ws;
  size_t o = 0;
  float* r   = f + o; o += BTC;
  float* k   = f + o; o += BTC;
  float* v   = f + o; o += BTC;
  float* zw  = f + o; o += BTC;     // w-path out -> decay (in place)
  float* za  = f + o; o += BTC;     // a-path out -> a_sig (in place)
  float* gg  = f + o; o += BTC;     // gate
  float* zv  = f + o; o += BTC;     // v-path out; reused as scan output y
  float* dd  = f + o; o += BTC;
  float* aa  = f + o; o += BTC;
  float* bb  = f + o; o += BTC;
  // union region: 6 f16 mix buffers, later reused as rr/kk2/vv (f32)
  float* uni = f + o; o += 3 * BTC;
  _Float16* xr16 = (_Float16*)uni;
  _Float16* xw16 = xr16 + BTC;     _Float16* xk16 = xr16 + 2 * BTC;
  _Float16* xv16 = xr16 + 3 * BTC; _Float16* xa16 = xr16 + 4 * BTC;
  _Float16* xg16 = xr16 + 5 * BTC;
  float* rr = uni; float* kk2 = uni + BTC; float* vv = uni + 2 * BTC;
  // f16 weights: big ones converted in place ([N,K] already), MLP ones
  // transposed+converted to [N,K] so every GEMM takes the B128 fast path.
  _Float16* Wr16 = (_Float16*)(f + o); o += CC / 2;
  _Float16* Wk16 = (_Float16*)(f + o); o += CC / 2;
  _Float16* Wv16 = (_Float16*)(f + o); o += CC / 2;
  _Float16* Wo16 = (_Float16*)(f + o); o += CC / 2;
  _Float16* w1t = (_Float16*)(f + o); o += (size_t)RW_C * 96 / 2;   // [96,2048]
  _Float16* w2t = (_Float16*)(f + o); o += (size_t)RW_C * 96 / 2;   // [2048,96]
  _Float16* a1t = (_Float16*)(f + o); o += (size_t)RW_C * 96 / 2;
  _Float16* a2t = (_Float16*)(f + o); o += (size_t)RW_C * 96 / 2;
  _Float16* v1t = (_Float16*)(f + o); o += (size_t)RW_C * 64 / 2;
  _Float16* v2t = (_Float16*)(f + o); o += (size_t)RW_C * 64 / 2;
  _Float16* g1t = (_Float16*)(f + o); o += (size_t)RW_C * 256 / 2;
  _Float16* g2t = (_Float16*)(f + o); o += (size_t)RW_C * 256 / 2;
  _Float16* y216 = (_Float16*)(f + o); o += BTC / 2;
  // f32 hidden + f16 hidden
  float* hw = f + o; o += (size_t)RW_M * 96;
  float* ha = f + o; o += (size_t)RW_M * 96;
  float* hv = f + o; o += (size_t)RW_M * 64;
  float* hg = f + o; o += (size_t)RW_M * 256;
  _Float16* hw16 = (_Float16*)(f + o); o += (size_t)RW_M * 96 / 2;
  _Float16* ha16 = (_Float16*)(f + o); o += (size_t)RW_M * 96 / 2;
  _Float16* hv16 = (_Float16*)(f + o); o += (size_t)RW_M * 64 / 2;
  _Float16* hg16 = (_Float16*)(f + o); o += (size_t)RW_M * 256 / 2;
  float* ybuf = zv;                  // zv dead after k_vmix

  // second tuple element: v_first passthrough
  hipMemcpyAsync(out + BTC, vf, BTC * sizeof(float), hipMemcpyDeviceToDevice, stream);

  // big projection weights: already [N,K]; straight convert
  auto cvt = [&](const float* src, _Float16* dst, size_t n) {
    k_cvt<<<(int)((n + 255) / 256), 256, 0, stream>>>(src, dst, (int)n);
  };
  cvt(W_r, Wr16, CC); cvt(W_k, Wk16, CC); cvt(W_v, Wv16, CC); cvt(W_o, Wo16, CC);
  // MLP weights: transpose [K,N] -> [N,K] + convert
  auto tcvt = [&](const float* src, _Float16* dst, int K, int N) {
    k_tcvt<<<dim3(N / 32, K / 32), dim3(32, 8), 0, stream>>>(src, dst, K, N);
  };
  tcvt(w1, w1t, RW_C, 96);  tcvt(w2, w2t, 96, RW_C);
  tcvt(a1, a1t, RW_C, 96);  tcvt(a2, a2t, 96, RW_C);
  tcvt(v1, v1t, RW_C, 64);  tcvt(v2, v2t, 64, RW_C);
  tcvt(g1, g1t, RW_C, 256); tcvt(g2, g2t, 256, RW_C);

  const int nb = (int)((BTC + 255) / 256);
  k_mix<<<nb, 256, 0, stream>>>(x, mr, mw, mk, mv, ma, mg,
                                xr16, xw16, xk16, xv16, xa16, xg16, (int)BTC);

  const dim3 gBig(RW_C / 64, RW_M / 128);            // (32,32)
  gemm_wmma16<<<gBig, 256, 0, stream>>>(xr16, Wr16, r, RW_M, RW_C, RW_C);
  gemm_wmma16<<<gBig, 256, 0, stream>>>(xk16, Wk16, k, RW_M, RW_C, RW_C);
  gemm_wmma16<<<gBig, 256, 0, stream>>>(xv16, Wv16, v, RW_M, RW_C, RW_C);

  // w path: tanh(xw@w1)@w2 -> decay
  gemm_wmma16<<<dim3(2, RW_M / 128), 256, 0, stream>>>(xw16, w1t, hw, RW_M, 96, RW_C);
  k_tanh_h<<<(RW_M * 96 + 255) / 256, 256, 0, stream>>>(hw, hw16, RW_M * 96);
  gemm_wmma16<<<gBig, 256, 0, stream>>>(hw16, w2t, zw, RW_M, RW_C, 96);
  k_decay<<<nb, 256, 0, stream>>>(zw, w0, (int)BTC);

  // a path: sigmoid(a0 + (xa@a1)@a2)
  gemm_wmma16<<<dim3(2, RW_M / 128), 256, 0, stream>>>(xa16, a1t, ha, RW_M, 96, RW_C);
  k_cvt<<<(RW_M * 96 + 255) / 256, 256, 0, stream>>>(ha, ha16, RW_M * 96);
  gemm_wmma16<<<gBig, 256, 0, stream>>>(ha16, a2t, za, RW_M, RW_C, 96);
  k_sigb<<<nb, 256, 0, stream>>>(za, a0, (int)BTC);

  // v path: v += (v_first - v)*sigmoid(v0 + (xv@v1)@v2)
  gemm_wmma16<<<dim3(1, RW_M / 128), 256, 0, stream>>>(xv16, v1t, hv, RW_M, 64, RW_C);
  k_cvt<<<(RW_M * 64 + 255) / 256, 256, 0, stream>>>(hv, hv16, RW_M * 64);
  gemm_wmma16<<<gBig, 256, 0, stream>>>(hv16, v2t, zv, RW_M, RW_C, 64);
  k_vmix<<<nb, 256, 0, stream>>>(v, vf, zv, v0, (int)BTC);

  // g path: sigmoid(xg@g1)@g2
  gemm_wmma16<<<dim3(4, RW_M / 128), 256, 0, stream>>>(xg16, g1t, hg, RW_M, 256, RW_C);
  k_sig_h<<<(RW_M * 256 + 255) / 256, 256, 0, stream>>>(hg, hg16, RW_M * 256);
  gemm_wmma16<<<gBig, 256, 0, stream>>>(hg16, g2t, gg, RW_M, RW_C, 256);

  // scan prep -> scan -> groupnorm/bonus/gate
  k_prep<<<RW_M, 256, 0, stream>>>(r, k, v, zw, za, k_k, k_a, rr, kk2, vv, dd, aa, bb);
  k_scan<<<RW_B * RW_H, 64, 0, stream>>>(rr, kk2, vv, dd, aa, bb, ybuf);
  k_post<<<RW_M, 256, 0, stream>>>(ybuf, rr, kk2, vv, gg, r_k, lng, lnb, y216);

  // final projection
  gemm_wmma16<<<gBig, 256, 0, stream>>>(y216, Wo16, out, RW_M, RW_C, RW_C);
}
